// Export_83614423318727
// MI455X (gfx1250) — compile-verified, matrix operand order checked by reading, not run on hardware
//
#include <hip/hip_runtime.h>
#include <stdint.h>

// ============================================================================
// CenterNet decode (sigmoid + 3x3 maxpool peaks + per-stage top-100) + greedy
// NMS for MI455X (gfx1250, wave32).
//
// Roofline: only the 2x2 heat planes (16.8MB) must stream from HBM; second
// pass hits 192MB L2. ~1-2us floor at 23.3TB/s. No GEMM exists in the
// reference, so WMMA is used only where a real (small) matmul appears: the
// stable argsort of 200 boxes as a 0/1-comparison-matrix row-sum via
// v_wmma_i32_16x16x64_iu8. The hot maxpool kernel stages halo tiles into LDS
// with CDNA5 async global->LDS copies (ASYNCcnt) when the builtins exist.
// ============================================================================

typedef unsigned int u32;
typedef unsigned long long u64;
typedef int v8i __attribute__((ext_vector_type(8)));

#define HDIM 1024
#define WDIM 1024
#define HW (HDIM * WDIM)
#define TOPK 100
#define NBOX 200
#define SCALE_F 4.0f
#define CONF_THF 0.3f
#define NMS_THF 0.5f
#define NBINS 32768
#define CAND_MAX 4096

// ---- optional CDNA5 async global->LDS path (guarded; clean fallback) -------
#if defined(__has_builtin)
#if __has_builtin(__builtin_amdgcn_global_load_async_to_lds_b32) && \
    __has_builtin(__builtin_amdgcn_s_wait_asynccnt)
#define USE_ASYNC_LDS 1
typedef __attribute__((address_space(1))) int* glb_i32p;
typedef __attribute__((address_space(3))) int* lds_i32p;
#endif
#endif

__device__ __forceinline__ float sigmoidf_(float v) {
  return 1.0f / (1.0f + expf(-v));
}

// ---------------------------------------------------------------------------
// k_init: zero histogram + counters at the start of every launch (ws state
// must not leak across graph replays).
// ---------------------------------------------------------------------------
__global__ void k_init(u32* __restrict__ p, int n) {
  int i = blockIdx.x * 256 + threadIdx.x;
  if (i < n) p[i] = 0u;
}

// ---------------------------------------------------------------------------
// k_peak: per (stage, class, 32x32 tile): stage 34x34 halo of logits into LDS
// (async-to-LDS when available), sigmoid in place, 3x3 max. A pixel is a peak
// iff its sigmoid equals the window max (matches hmax==heat; -inf padding at
// image borders matches reduce_window 'SAME' init).
//   pass 0: 15-bit float-bits histogram of nonzero peak values (global atomics)
//   pass 1: compact keys (bits<<32 | ~idx) for bins >= per-stage threshold
// ---------------------------------------------------------------------------
__global__ __launch_bounds__(256) void k_peak(const float* __restrict__ x,
                                              u32* __restrict__ hist,
                                              u32* __restrict__ candCnt,
                                              const u32* __restrict__ thr,
                                              u64* __restrict__ candKeys,
                                              int pass) {
  __shared__ float tile[34][36];  // 36 stride dodges bank conflicts
  const int tid = threadIdx.x;
  const int ty = blockIdx.x >> 5, tx = blockIdx.x & 31;
  const int c = blockIdx.y, s = blockIdx.z;
  const float* plane = x + (size_t)(s * 6 + c) * HW;
  const int gy0 = ty * 32 - 1, gx0 = tx * 32 - 1;

  for (int i = tid; i < 34 * 34; i += 256) {
    int ly = i / 34, lx = i - ly * 34;
    int gy = gy0 + ly, gx = gx0 + lx;
    bool in = (gy >= 0) & (gy < HDIM) & (gx >= 0) & (gx < WDIM);
#if defined(USE_ASYNC_LDS)
    if (in) {
      __builtin_amdgcn_global_load_async_to_lds_b32(
          (glb_i32p)(plane + (size_t)gy * WDIM + gx),
          (lds_i32p)&tile[ly][lx], 0, 0);
    } else {
      tile[ly][lx] = -__builtin_inff();
    }
#else
    tile[ly][lx] = in ? plane[(size_t)gy * WDIM + gx] : -__builtin_inff();
#endif
  }
#if defined(USE_ASYNC_LDS)
  __builtin_amdgcn_s_wait_asynccnt(0);
#endif
  __syncthreads();
  for (int i = tid; i < 34 * 34; i += 256) {
    int ly = i / 34, lx = i - ly * 34;
    tile[ly][lx] = sigmoidf_(tile[ly][lx]);  // sigmoid(-inf) -> 0, harmless for max
  }
  __syncthreads();

  const u32 Ts = (pass == 1) ? thr[s] : 0u;
  for (int k = 0; k < 4; ++k) {
    int o = tid + k * 256;
    int oy = o >> 5, ox = o & 31;
    float hc = tile[oy + 1][ox + 1];
    float m = hc;
#pragma unroll
    for (int dy = 0; dy < 3; ++dy)
#pragma unroll
      for (int dx = 0; dx < 3; ++dx) m = fmaxf(m, tile[oy + dy][ox + dx]);
    if (m == hc) {  // peak; hc > 0 always (sigmoid)
      u32 bits = __float_as_uint(hc);
      u32 bin = bits >> 16;
      if (pass == 0) {
        atomicAdd(&hist[(size_t)s * NBINS + bin], 1u);
      } else if (bin >= Ts) {
        u32 pos = atomicAdd(&candCnt[s], 1u);
        if (pos < CAND_MAX) {
          u32 idx = (u32)c * HW + (u32)((ty * 32 + oy) * WDIM + (tx * 32 + ox));
          candKeys[(size_t)s * CAND_MAX + pos] =
              ((u64)bits << 32) | (u64)(0xFFFFFFFFu - idx);
        }
      }
    }
  }
}

// ---------------------------------------------------------------------------
// k_thresh: per stage, find the largest bin T with suffix-count >= TOPK.
// 256 threads each suffix-sum a 128-bin chunk, then thread 0 refines.
// ---------------------------------------------------------------------------
__global__ __launch_bounds__(256) void k_thresh(const u32* __restrict__ hist,
                                                u32* __restrict__ thr) {
  const int s = blockIdx.x, tid = threadIdx.x;
  const u32* h = hist + (size_t)s * NBINS;
  __shared__ u32 csum[256];
  u32 sum = 0;
  int hi = NBINS - 1 - tid * 128;
  for (int k = 0; k < 128; ++k) sum += h[hi - k];
  csum[tid] = sum;
  __syncthreads();
  if (tid == 0) {
    u32 cum = 0, T = 0;
    for (int t = 0; t < 256; ++t) {
      if (cum + csum[t] >= (u32)TOPK) {
        int hb = NBINS - 1 - t * 128;
        for (int k = 0; k < 128; ++k) {
          cum += h[hb - k];
          if (cum >= (u32)TOPK) { T = (u32)(hb - k); break; }
        }
        break;
      }
      cum += csum[t];
    }
    thr[s] = T;  // T==0 only if fewer than TOPK positive peaks exist
  }
}

// ---------------------------------------------------------------------------
// k_topk: one block per stage. Exact top-100 over <=4096 candidate keys in
// LDS (keys are distinct: value bits + inverted index => matches lax.top_k
// ordering incl. ties). Decode boxes / class / thresholded score.
// ---------------------------------------------------------------------------
__global__ __launch_bounds__(256) void k_topk(const float* __restrict__ x,
                                              const u64* __restrict__ candKeys,
                                              const u32* __restrict__ candCnt,
                                              float* __restrict__ boxes,
                                              float* __restrict__ clsArr,
                                              float* __restrict__ scArr) {
  __shared__ u64 keys[CAND_MAX];
  __shared__ u64 red[256];
  const int s = blockIdx.x, tid = threadIdx.x;
  u32 nc = candCnt[s];
  if (nc > CAND_MAX) nc = CAND_MAX;
  for (int i = tid; i < CAND_MAX; i += 256)
    keys[i] = (i < (int)nc) ? candKeys[(size_t)s * CAND_MAX + i] : 0ull;
  __syncthreads();

  for (int r = 0; r < TOPK; ++r) {
    u64 m = 0ull;
    for (int i = tid; i < CAND_MAX; i += 256) {
      u64 k2 = keys[i];
      if (k2 > m) m = k2;
    }
    red[tid] = m;
    __syncthreads();
    for (int st = 128; st > 0; st >>= 1) {
      if (tid < st && red[tid + st] > red[tid]) red[tid] = red[tid + st];
      __syncthreads();
    }
    u64 kmax = red[0];
    for (int i = tid; i < CAND_MAX; i += 256)
      if (keys[i] == kmax) keys[i] = 0ull;  // unique keys -> removes the winner
    if (tid == 0) {
      float val = __uint_as_float((u32)(kmax >> 32));
      u32 idx = 0xFFFFFFFFu - (u32)(kmax & 0xFFFFFFFFull);
      u32 cls = idx / HW;
      u32 pix = idx - cls * HW;
      float ysf = (float)(pix / WDIM);
      float xsf = (float)(pix % WDIM);
      const float* base = x + (size_t)s * 6 * HW;
      float offx = base[(size_t)2 * HW + pix];
      float offy = base[(size_t)3 * HW + pix];
      float wv = base[(size_t)4 * HW + pix];
      float hv = base[(size_t)5 * HW + pix];
      float cx = xsf + offx, cy = ysf + offy;
      float hw2 = wv * 0.5f, hh2 = hv * 0.5f;
      int o = s * TOPK + r;
      boxes[o * 4 + 0] = (cx - hw2) * SCALE_F;
      boxes[o * 4 + 1] = (cy - hh2) * SCALE_F;
      boxes[o * 4 + 2] = (cx + hw2) * SCALE_F;
      boxes[o * 4 + 3] = (cy + hh2) * SCALE_F;
      clsArr[o] = (float)cls;
      scArr[o] = (val > CONF_THF) ? val : 0.0f;
    }
    __syncthreads();
  }
}

// ---------------------------------------------------------------------------
// k_nms: single block.
//   1) stable descending argsort of 200 scores: rank[i] = #{j: key_j>key_i}
//      computed as a 0/1 matrix row-sum with v_wmma_i32_16x16x64_iu8
//      (13 row-tiles x 4 k-tiles over 8 waves; B = all-ones so every D column
//      holds the row sum; A laid out per ISA 8-bit A-matrix table).
//   2) gather sorted boxes/scores/classes, compute areas.
//   3) faithful sequential greedy NMS (199 steps, parallel over j).
// ---------------------------------------------------------------------------
__global__ __launch_bounds__(256) void k_nms(const float* __restrict__ boxesIn,
                                             const float* __restrict__ clsIn,
                                             const float* __restrict__ scIn,
                                             float* __restrict__ out) {
  __shared__ u64 keys[256];
  __shared__ int rnk[256];
  __shared__ float bx1[NBOX], by1[NBOX], bx2[NBOX], by2[NBOX];
  __shared__ float area[NBOX], sc[NBOX], co[NBOX];
  const int tid = threadIdx.x;
  float myScore = 0.0f;
  keys[tid] = 0ull;  // pads (>=NBOX) are 0 => contribute 0 to every rank
  if (tid < NBOX) {
    myScore = scIn[tid];
    keys[tid] = ((u64)__float_as_uint(myScore) << 32) |
                (u64)(0xFFFFFFFFu - (u32)tid);  // ties -> smaller index first
  }
  __syncthreads();

  const int wave = tid >> 5, lane = tid & 31;
  const int basek[8] = {0, 4, 16, 20, 32, 36, 48, 52};  // ISA 8-bit A layout
  const int hi8 = (lane >= 16) ? 8 : 0;
  v8i bones;
#pragma unroll
  for (int e = 0; e < 8; ++e) bones[e] = 0x01010101;

#pragma unroll
  for (int pass = 0; pass < 2; ++pass) {
    if (pass == 1 && wave >= 5) break;  // tiles 0..12 cover rows 0..207
    const int t = (pass == 0) ? wave : (8 + wave);
    const int m = t * 16 + (lane & 15);
    const u64 km = keys[m];
    v8i acc;
#pragma unroll
    for (int e = 0; e < 8; ++e) acc[e] = 0;
#pragma unroll
    for (int kt = 0; kt < 4; ++kt) {
      v8i a;
#pragma unroll
      for (int v = 0; v < 8; ++v) {
        int K = kt * 64 + basek[v] + hi8;
        u32 byt = 0;
#pragma unroll
        for (int b2 = 0; b2 < 4; ++b2)
          byt |= (keys[K + b2] > km ? 1u : 0u) << (8 * b2);
        a[v] = (int)byt;
      }
      acc = __builtin_amdgcn_wmma_i32_16x16x64_iu8(false, a, false, bones, acc,
                                                   false, false);
    }
    // D layout: VGPR r holds M=r (lanes 0-15) / M=8+r (lanes 16-31); all N equal.
    if (lane == 0 || lane == 16) {
      int mb = t * 16 + ((lane == 16) ? 8 : 0);
#pragma unroll
      for (int r = 0; r < 8; ++r) {
        int row = mb + r;
        if (row < NBOX) rnk[row] = acc[r];
      }
    }
  }
  __syncthreads();

  if (tid < NBOX) {  // scatter into sorted order (ranks are a permutation)
    int r = rnk[tid];
    bx1[r] = boxesIn[tid * 4 + 0];
    by1[r] = boxesIn[tid * 4 + 1];
    bx2[r] = boxesIn[tid * 4 + 2];
    by2[r] = boxesIn[tid * 4 + 3];
    sc[r] = myScore;
    co[r] = clsIn[tid];
  }
  __syncthreads();
  if (tid < NBOX)
    area[tid] = (bx2[tid] - bx1[tid] + 1.0f) * (by2[tid] - by1[tid] + 1.0f);
  __syncthreads();

  for (int i = 0; i < NBOX - 1; ++i) {
    float si = sc[i];
    if (tid > i && tid < NBOX && si > 0.0f) {
      float xx1 = fmaxf(bx1[i], bx1[tid]);
      float yy1 = fmaxf(by1[i], by1[tid]);
      float xx2 = fminf(bx2[i], bx2[tid]);
      float yy2 = fminf(by2[i], by2[tid]);
      float inter = fmaxf(xx2 - xx1 + 1.0f, 0.0f) * fmaxf(yy2 - yy1 + 1.0f, 0.0f);
      float iou = inter / (area[i] + area[tid] - inter);
      if (iou >= NMS_THF) sc[tid] = 0.0f;
    }
    __syncthreads();
  }

  if (tid < NBOX) {  // output: b_sorted (200x4), clss[order], s_final
    out[tid * 4 + 0] = bx1[tid];
    out[tid * 4 + 1] = by1[tid];
    out[tid * 4 + 2] = bx2[tid];
    out[tid * 4 + 3] = by2[tid];
    out[4 * NBOX + tid] = co[tid];
    out[5 * NBOX + tid] = sc[tid];
  }
}

// ---------------------------------------------------------------------------
// Workspace layout (bytes): hist 2*32768*4 = 262144 | candCnt 8 | thr 8 |
// pad to 262176 | candKeys 2*4096*8 = 65536 (-> 327712) | boxes 3200 |
// cls 800 | scores 800. Total ~333 KB.
// ---------------------------------------------------------------------------
extern "C" void kernel_launch(void* const* d_in, const int* in_sizes, int n_in,
                              void* d_out, int out_size, void* d_ws,
                              size_t ws_size, hipStream_t stream) {
  (void)in_sizes; (void)n_in; (void)out_size; (void)ws_size;
  const float* xin = (const float*)d_in[0];
  u32* hist = (u32*)d_ws;
  u32* candCnt = hist + 2 * NBINS;
  u32* thr = candCnt + 2;
  u64* candKeys = (u64*)((char*)d_ws + 262176);
  float* boxes = (float*)((char*)d_ws + 327712);
  float* clsArr = boxes + NBOX * 4;
  float* scArr = clsArr + NBOX;

  const int nzero = 2 * NBINS + 4;
  k_init<<<(nzero + 255) / 256, 256, 0, stream>>>((u32*)d_ws, nzero);
  k_peak<<<dim3(1024, 2, 2), 256, 0, stream>>>(xin, hist, candCnt, thr,
                                               candKeys, 0);
  k_thresh<<<2, 256, 0, stream>>>(hist, thr);
  k_peak<<<dim3(1024, 2, 2), 256, 0, stream>>>(xin, hist, candCnt, thr,
                                               candKeys, 1);
  k_topk<<<2, 256, 0, stream>>>(xin, candKeys, candCnt, boxes, clsArr, scArr);
  k_nms<<<1, 256, 0, stream>>>(boxes, clsArr, scArr, (float*)d_out);
}
